// LSTMCell_53300544143475
// MI455X (gfx1250) — matrix operand order repaired and checked by reading, not compiled, not run on hardware
//
#include <hip/hip_runtime.h>

#define B_DIM  4096
#define IN_DIM 1024
#define H_DIM  1024
#define K_DIM  (IN_DIM + H_DIM)

typedef __attribute__((ext_vector_type(16))) __bf16 bf16x16;
typedef __attribute__((ext_vector_type(8)))  __bf16 bf16x8;
typedef __attribute__((ext_vector_type(8)))  float  f32x8;

union Frag16 { bf16x16 v; bf16x8 h[2]; };

__device__ __forceinline__ float fast_sigmoid(float xv) {
    return 1.0f / (1.0f + __expf(-xv));
}
__device__ __forceinline__ float fast_tanh(float xv) {
    // tanh(x) = 1 - 2/(exp(2x)+1); saturates to +/-1 without inf/inf NaN
    return 1.0f - 2.0f / (__expf(2.0f * xv) + 1.0f);
}
__device__ __forceinline__ bf16x8 cvt8(float4 lo, float4 hi) {
    bf16x8 r;
    r[0] = (__bf16)lo.x; r[1] = (__bf16)lo.y; r[2] = (__bf16)lo.z; r[3] = (__bf16)lo.w;
    r[4] = (__bf16)hi.x; r[5] = (__bf16)hi.y; r[6] = (__bf16)hi.z; r[7] = (__bf16)hi.w;
    return r;
}

// Register staging for one K=32 slab (fetched one iteration ahead).
struct Stage {
    float4 a[4];   // 16 floats of the combined [x|h_prev] row
    float4 w[4];   // 16 floats of one W row chunk
};

// Fused LSTM cell, WMMA bf16 GEMM core.
// WG tile: M=128 (batch) x N=32 (hidden) x all 4 gates.
// 8 wave32 waves as 4(M) x 2(N); each wave owns two 16-row M fragments,
// so every B fragment read from LDS feeds 2 WMMAs (12 ds_loads : 8 wmma).
__global__ __launch_bounds__(256)
void lstm_cell_wmma_bf16(const float* __restrict__ x,
                         const float* __restrict__ h_prev,
                         const float* __restrict__ c_prev,
                         const float* __restrict__ W,
                         const float* __restrict__ bias,
                         float* __restrict__ out)
{
    // Row stride padded 32->40 bf16 (80B: multiple of 16B keeps ds_*_b128 legal,
    // and 80B row strides walk banks conflict-free for 16-row fragment reads).
    __shared__ __align__(16) __bf16 lA[128][40];       // A tile: 128 rows x 32 K (10 KB)
    __shared__ __align__(16) __bf16 lB[4][32][40];     // W tiles: gate x 32 N x 32 K (10 KB)

    const int tid  = threadIdx.x;
    const int lane = tid & 31;
    const int wave = tid >> 5;
    const int wm   = wave & 3;           // wave M position (0..3) -> 32 rows each
    const int wn   = wave >> 2;          // wave N position (0..1)
    const int mBase = blockIdx.y * 128;  // batch-row base
    const int nBase = blockIdx.x * 32;   // hidden-col base

    // Cooperative staging assignments: 256 threads, 16 f32 each per tile.
    const int aRow  = tid >> 1;          // 0..127
    const int aCol0 = (tid & 1) * 16;    // 0,16
    const int bGate = tid >> 6;          // 0..3
    const int bN    = (tid >> 1) & 31;   // 0..31
    const int bCol0 = (tid & 1) * 16;    // 0,16

    // Fragment coordinates (ISA 7.12.2 wave32 layouts).
    const int mo = wm * 32;              // two 16-row fragments: mo, mo+16
    const int am = lane & 15;
    const int ak = (lane >> 4) * 8;      // A: K-half base (0/8); +16 for upper 8 elems
    const int bn = wn * 16 + (lane & 15);
    const int bk = (lane >> 4) * 8;      // B: K-half base (0/8); +16 for upper 8 elems

    f32x8 acc0[4] = {{}, {}, {}, {}};    // gate accumulators, rows mo..mo+15
    f32x8 acc1[4] = {{}, {}, {}, {}};    // gate accumulators, rows mo+16..mo+31

    const size_t aRowOffX = (size_t)(mBase + aRow) * IN_DIM;
    const size_t aRowOffH = (size_t)(mBase + aRow) * H_DIM;
    const size_t wRowOff  = (size_t)(bGate * H_DIM + nBase + bN) * K_DIM;

    auto load_stage = [&](Stage& s, int kc) {
        const float* srcA = (kc < IN_DIM)
            ? (x      + aRowOffX + kc            + aCol0)
            : (h_prev + aRowOffH + (kc - IN_DIM) + aCol0);
#pragma unroll
        for (int i = 0; i < 4; ++i) s.a[i] = ((const float4*)srcA)[i];
        const float* srcW = W + wRowOff + kc + bCol0;
#pragma unroll
        for (int i = 0; i < 4; ++i) s.w[i] = ((const float4*)srcW)[i];
    };

    Stage cur, nxt;
    load_stage(cur, 0);

    for (int kc = 0; kc < K_DIM; kc += 32) {
        __syncthreads();   // previous iteration's fragment readers are done
        *(bf16x8*)&lA[aRow][aCol0]          = cvt8(cur.a[0], cur.a[1]);
        *(bf16x8*)&lA[aRow][aCol0 + 8]      = cvt8(cur.a[2], cur.a[3]);
        *(bf16x8*)&lB[bGate][bN][bCol0]     = cvt8(cur.w[0], cur.w[1]);
        *(bf16x8*)&lB[bGate][bN][bCol0 + 8] = cvt8(cur.w[2], cur.w[3]);
        __syncthreads();   // tiles visible

        // Prefetch next slab while WMMAs run (wait lands under the matrix ops).
        if (kc + 32 < K_DIM) load_stage(nxt, kc + 32);

        Frag16 a0, a1;
        a0.h[0] = *(const bf16x8*)&lA[mo + am][ak];
        a0.h[1] = *(const bf16x8*)&lA[mo + am][ak + 16];
        a1.h[0] = *(const bf16x8*)&lA[mo + 16 + am][ak];
        a1.h[1] = *(const bf16x8*)&lA[mo + 16 + am][ak + 16];

#pragma unroll
        for (int g = 0; g < 4; ++g) {
            Frag16 bfr;
            bfr.h[0] = *(const bf16x8*)&lB[g][bn][bk];
            bfr.h[1] = *(const bf16x8*)&lB[g][bn][bk + 16];
            acc0[g] = __builtin_amdgcn_wmma_f32_16x16x32_bf16(false, a0.v, false, bfr.v,
                                                              (short)0, acc0[g], false, false);
            acc1[g] = __builtin_amdgcn_wmma_f32_16x16x32_bf16(false, a1.v, false, bfr.v,
                                                              (short)0, acc1[g], false, false);
        }

        cur = nxt;
    }

    // ---- fused epilogue: bias + activations + LSTM state update
    const int nG = nBase + bn;                       // global hidden column
    float bgate[4];
#pragma unroll
    for (int g = 0; g < 4; ++g) bgate[g] = bias[g * H_DIM + nG];

    const int mTop = mBase + mo + (lane >> 4) * 8;   // C/D layout: vgpr r -> row mTop+r

    float* __restrict__ hOut = out;
    float* __restrict__ cOut = out + (size_t)B_DIM * H_DIM;

#pragma unroll
    for (int half = 0; half < 2; ++half) {
        const f32x8* acc = half ? acc1 : acc0;
        const int mh = mTop + half * 16;
#pragma unroll
        for (int r = 0; r < 8; ++r) {
            const size_t idx = (size_t)(mh + r) * H_DIM + nG;
            const float fg = fast_sigmoid(acc[0][r] + bgate[0]);
            const float ig = fast_sigmoid(acc[1][r] + bgate[1]);
            const float gg = fast_tanh   (acc[2][r] + bgate[2]);
            const float og = fast_sigmoid(acc[3][r] + bgate[3]);
            const float cp = c_prev[idx];
            const float ct = fg * cp + ig * gg;
            const float ht = og * fast_tanh(ct);
            hOut[idx] = ht;
            cOut[idx] = ct;
        }
    }
}

extern "C" void kernel_launch(void* const* d_in, const int* in_sizes, int n_in,
                              void* d_out, int out_size, void* d_ws, size_t ws_size,
                              hipStream_t stream) {
    (void)in_sizes; (void)n_in; (void)out_size; (void)d_ws; (void)ws_size;
    const float* x      = (const float*)d_in[0];
    const float* h_prev = (const float*)d_in[1];
    const float* c_prev = (const float*)d_in[2];
    const float* W      = (const float*)d_in[3];
    const float* bias   = (const float*)d_in[4];
    float* out          = (float*)d_out;

    dim3 grid(H_DIM / 32, B_DIM / 128);  // 32 N-blocks x 32 M-blocks = 1024 WGs
    dim3 block(256);                     // 8 wave32 waves: 4(M) x 2(N)
    lstm_cell_wmma_bf16<<<grid, block, 0, stream>>>(x, h_prev, c_prev, W, bias, out);
}
